// CustomLSTM_11458972746257
// MI455X (gfx1250) — matrix operand order repaired and checked by reading, not compiled, not run on hardware
//
#include <hip/hip_runtime.h>
#include <hip/hip_bf16.h>

typedef __attribute__((ext_vector_type(16))) _Float16 v16h;
typedef __attribute__((ext_vector_type(8)))  _Float16 v8h;
typedef __attribute__((ext_vector_type(8)))  float    v8f;

#define T_STEPS 262144
#define I_DIM   6
#define H_DIM   50
#define G_DIM   200      // 4*H
#define K_TOT   56       // H + I, padded to 64 by two K=32 WMMA slices
#define N_WAVES 13       // ceil(200/16)
#define SCAN_BLK (N_WAVES*32)
#define CH      32       // steps per x-staging chunk
#define NBLK    64       // LDS A-block ring depth (2 chunks)

// ---- gfx1250 hardware transcendentals -------------------------------------
#if defined(__has_builtin) && __has_builtin(__builtin_amdgcn_tanhf)
__device__ __forceinline__ float tanh_hw(float x) {   // v_tanh_f32
  return __builtin_amdgcn_tanhf(x);
}
#else
__device__ __forceinline__ float tanh_hw(float x) {   // v_exp + v_rcp, no IEEE div
  float e = __expf(2.f * x);
  return __fmaf_rn(-2.f, __builtin_amdgcn_rcpf(e + 1.f), 1.f);
}
#endif

__device__ __forceinline__ float sigmoid_hw(float x) {
  return __builtin_amdgcn_rcpf(1.f + __expf(-x));     // v_exp + v_add + v_rcp
}

// ---- LDS-only workgroup barrier: only waits dscnt, never drains the
// fire-and-forget hs stores or the chunk-staging global loads.
__device__ __forceinline__ void wg_sync_lds() {
#if defined(__has_builtin) && __has_builtin(__builtin_amdgcn_s_barrier_signal) && \
    __has_builtin(__builtin_amdgcn_s_barrier_wait)
  __builtin_amdgcn_fence(__ATOMIC_RELEASE, "workgroup", "local"); // s_wait_dscnt 0
  __builtin_amdgcn_s_barrier_signal(-1);
  __builtin_amdgcn_s_barrier_wait((short)-1);
  __builtin_amdgcn_fence(__ATOMIC_ACQUIRE, "workgroup", "local");
#else
  __syncthreads();
#endif
}

// Fetch combined weight matrix element B[k][j] = [W_hh | W_ih](j, k), as f16.
__device__ __forceinline__ _Float16 wfetch(int j, int k, const float* __restrict__ Whh,
                                           const float* __restrict__ Wih) {
  if (j >= G_DIM) return (_Float16)0.f;
  if (k < H_DIM)  return (_Float16)Whh[j * H_DIM + k];
  if (k < K_TOT)  return (_Float16)Wih[j * I_DIM + (k - H_DIM)];
  return (_Float16)0.f;
}

// Persistent single-workgroup LSTM scan. One wave owns 16 gate columns.
// gates(t) = [h(t-1) | x(t)] @ [W_hh | W_ih]^T + (b_ih + b_hh), done as two
// chained v_wmma_f32_16x16x32_f16 (K=0..31 and K=32..63, K>=56 zero padded).
// A-operands live in a 64-deep LDS ring of 64-half blocks; x(t) components are
// staged a full chunk (32 steps) ahead so global-load latency never hits the
// serial chain.
__global__ __launch_bounds__(SCAN_BLK) void lstm_scan_kernel(
    const float* __restrict__ inSeq,   // [T, 6]
    const float* __restrict__ W_ih,    // [200, 6]
    const float* __restrict__ W_hh,    // [200, 50]
    const float* __restrict__ b_ih,    // [200]
    const float* __restrict__ b_hh,    // [200]
    float* __restrict__ hs)            // [T, 50] workspace
{
  __shared__ __align__(16) _Float16 hbuf[NBLK * 64]; // block b: h | x | zeros
  __shared__ float act[G_DIM + 8];                   // activated gates i,f,g,o

  const int tid  = threadIdx.x;
  const int lane = tid & 31;
  const int wv   = tid >> 5;
  const int col  = wv * 16 + (lane & 15);      // gate index this lane produces

  // ---- Resident B operand: contiguous k order, same as A (32x16 per slice, 8 VGPRs):
  // lane n<16 holds (K=e, N=n), lane n>=16 holds (K=16+e, N=n-16), e=0..15.
  v16h b0, b1;
  {
    const int kb = (lane < 16) ? 0 : 16;
#pragma unroll
    for (int e = 0; e < 16; ++e) {
      b0[e] = wfetch(col, kb + e,      W_hh, W_ih);
      b1[e] = wfetch(col, kb + e + 32, W_hh, W_ih);
    }
  }

  // ---- Branchless activation constants (per lane, set once):
  // act = off + outs*(0.5 + 0.5*tanh(ins2*(gate+bias)))
  //   sigmoid gates (i,f,o): ins2=0.5, outs=1, off=0  -> sigmoid(gate+bias)
  //   tanh gate (g):         ins2=1.0, outs=2, off=-1 -> tanh(gate+bias)
  const bool  is_g = (col >= 2 * H_DIM) && (col < 3 * H_DIM);
  const float ins2 = is_g ? 1.0f : 0.5f;
  const float outs = is_g ? 2.0f : 1.0f;
  const float offc = is_g ? -1.0f : 0.0f;
  const float bias = (col < G_DIM) ? (b_ih[col] + b_hh[col]) : 0.f;
  const float bb   = ins2 * bias;              // folded bias term

  // ---- init: zero the whole ring (h slots of block 0 = h(-1) = 0; pad slots
  // 56..63 of every block stay zero forever), then stage chunk 0's x values.
  for (int i = tid; i < NBLK * 64; i += SCAN_BLK) hbuf[i] = (_Float16)0.f;
  const int  sj     = tid;                       // staged x element (row*6+comp)
  const bool stager = (sj < CH * I_DIM);         // threads 0..191
  __syncthreads();
  if (stager)
    hbuf[(sj / I_DIM) * 64 + H_DIM + (sj % I_DIM)] = (_Float16)inSeq[sj];
  float cstate = 0.f;
  __syncthreads();

  const int NCHUNK = T_STEPS / CH;
  for (int chk = 0; chk < NCHUNK; ++chk) {
    // Issue next chunk's x loads now; LDS store happens ~30 steps later.
    float xstage = 0.f;
    const bool do_stage = stager && (chk + 1 < NCHUNK);
    if (do_stage) xstage = inSeq[(size_t)(chk + 1) * (CH * I_DIM) + sj];
    const int tbase = chk * CH;

    for (int s = 0; s < CH; ++s) {
      const int t   = tbase + s;
      const int blk = t & (NBLK - 1);
      const _Float16* __restrict__ ablk = hbuf + blk * 64;

      // ---- Build A (16x32 f16, identical vector broadcast into all 16 rows).
      // Half-wave-uniform: lanes<16 read K=e, lanes>=16 read K=16+e;
      // contiguous groups of 8 halves -> ds_load_b128 pairs.
      const int ab = (lane < 16) ? 0 : 16;
      v8h a0lo = *(const v8h*)(ablk + ab);
      v8h a0hi = *(const v8h*)(ablk + 8 + ab);
      v8h a1lo = *(const v8h*)(ablk + 32 + ab);
      v8h a1hi = *(const v8h*)(ablk + 40 + ab);
      v16h A0 = __builtin_shufflevector(a0lo, a0hi, 0,1,2,3,4,5,6,7,8,9,10,11,12,13,14,15);
      v16h A1 = __builtin_shufflevector(a1lo, a1hi, 0,1,2,3,4,5,6,7,8,9,10,11,12,13,14,15);

      // ---- Zero-init C (bias folded into activation), two chained WMMAs.
      v8f c = {0.f, 0.f, 0.f, 0.f, 0.f, 0.f, 0.f, 0.f};
      c = __builtin_amdgcn_wmma_f32_16x16x32_f16(false, A0, false, b0,
                                                 (short)0, c, false, false);
      c = __builtin_amdgcn_wmma_f32_16x16x32_f16(false, A1, false, b1,
                                                 (short)0, c, false, false);

      // All rows of D are identical -> c[0] on every lane is gate[col].
      const float targ = __fmaf_rn(ins2, c[0], bb);
      const float th   = tanh_hw(targ);
      const float av   = __fmaf_rn(outs, __fmaf_rn(0.5f, th, 0.5f), offc);
      if (lane < 16 && col < G_DIM) act[col] = av;
      wg_sync_lds();  // act ready; also fences A-reads before ring rewrites

      // ---- Elementwise cell update (threads 0..49 own one hidden unit).
      // h(t) goes into the NEXT step's A-block (slots 0..49).
      if (tid < H_DIM) {
        const float ig = act[tid];
        const float fg = act[H_DIM + tid];
        const float gg = act[2 * H_DIM + tid];
        const float og = act[3 * H_DIM + tid];
        cstate = __fmaf_rn(fg, cstate, ig * gg);
        const float hval = og * tanh_hw(cstate);
        hs[(size_t)t * H_DIM + tid] = hval;     // fire-and-forget global store
        hbuf[((t + 1) & (NBLK - 1)) * 64 + tid] = (_Float16)hval;
      }
      // ---- Late in the chunk: drop staged x(next chunk) into opposite-parity
      // blocks (slots 50..55). Load has had ~30 steps to complete.
      if (s == CH - 2 && do_stage) {
        const int tb = (tbase + CH + sj / I_DIM) & (NBLK - 1);
        hbuf[tb * 64 + H_DIM + (sj % I_DIM)] = (_Float16)xstage;
      }
      wg_sync_lds();  // ring block ready for next step
    }
  }
}

// pAction[t] = sigmoid(hs[t,:] . W_lin + b_lin) — trivially memory-bound.
__global__ void proj_kernel(const float* __restrict__ hs,
                            const float* __restrict__ W_lin,
                            const float* __restrict__ b_lin,
                            float* __restrict__ out)
{
  __shared__ float wl[H_DIM];
  const int tid = threadIdx.x;
  if (tid < H_DIM) wl[tid] = W_lin[tid];
  __syncthreads();

  const int t = blockIdx.x * blockDim.x + tid;  // T divisible by blockDim
  const float* __restrict__ hrow = hs + (size_t)t * H_DIM;
  float acc = b_lin[0];
#pragma unroll
  for (int k = 0; k < H_DIM; ++k) acc += hrow[k] * wl[k];
  out[t] = sigmoid_hw(acc);
}

extern "C" void kernel_launch(void* const* d_in, const int* in_sizes, int n_in,
                              void* d_out, int out_size, void* d_ws, size_t ws_size,
                              hipStream_t stream) {
  const float* inSeq = (const float*)d_in[0];  // [T,6]
  const float* W_ih  = (const float*)d_in[1];  // [200,6]
  const float* W_hh  = (const float*)d_in[2];  // [200,50]
  const float* b_ih  = (const float*)d_in[3];  // [200]
  const float* b_hh  = (const float*)d_in[4];  // [200]
  const float* W_lin = (const float*)d_in[5];  // [1,50]
  const float* b_lin = (const float*)d_in[6];  // [1]
  float* out = (float*)d_out;                  // [T]
  float* hs  = (float*)d_ws;                   // [T,50] = 52.4 MB scratch

  lstm_scan_kernel<<<1, SCAN_BLK, 0, stream>>>(inSeq, W_ih, W_hh, b_ih, b_hh, hs);
  proj_kernel<<<T_STEPS / 256, 256, 0, stream>>>(hs, W_lin, b_lin, out);
}
